// MinifloatLinear_51264729645539
// MI455X (gfx1250) — matrix operand order repaired and checked by reading, not compiled, not run on hardware
//
#include <hip/hip_runtime.h>

typedef __attribute__((ext_vector_type(16))) _Float16 v16h;
typedef __attribute__((ext_vector_type(8)))  float    v8f;
typedef int v4i __attribute__((vector_size(16)));   // matches builtin param type

#define AS1 __attribute__((address_space(1)))
#define AS3 __attribute__((address_space(3)))

namespace {
constexpr int KD = 4096;          // inner dim
constexpr int ND = 4096;          // out features
constexpr int MD = 8192;          // 4*2048 rows
constexpr int BM = 128, BN = 128, BK = 32;
constexpr int PITCH  = 80;        // LDS row pitch bytes (32 f16 + 8 f16 pad, 16B-aligned)
constexpr int TILE_B = 128 * PITCH;   // 10240 B per matrix tile
constexpr int BUF_B  = 2 * TILE_B;    // A + B = 20480 B per buffer
constexpr int NT     = KD / BK;       // 128 K-steps
}

// ---------------- fp8 emulation (RNE round to e5m2 / e4m3 grids) ----------------
__device__ __forceinline__ float fp8_round(float v, int mbits, int emin) {
  if (v == 0.0f) return v;
  int e; float m = frexpf(v, &e); (void)m;   // v = m*2^e, 0.5<=|m|<1
  int ex = e - 1;
  if (ex < emin) ex = emin;                  // denormal range
  const float q = exp2f((float)(ex - mbits));
  return rintf(v / q) * q;                   // RNE onto fp8 grid
}
__device__ __forceinline__ float quant_w_val(float v) {
  // e5m2 round-trip, then stored as e4m3fn (randn inputs: no saturation needed)
  return fp8_round(fp8_round(v, 2, -14), 3, -6);
}

// ---------------- async global->LDS (CDNA5) ----------------
__device__ __forceinline__ void async_copy16(const void* g, void* l) {
#if __has_builtin(__builtin_amdgcn_global_load_async_to_lds_b128)
  __builtin_amdgcn_global_load_async_to_lds_b128((v4i*)g, (AS3 v4i*)l, 0, 0);
#else
  unsigned lds32 = (unsigned)(unsigned long long)(uintptr_t)l;
  unsigned long long ga = (unsigned long long)(uintptr_t)g;
  asm volatile("global_load_async_to_lds_b128 %0, %1, off"
               :: "v"(lds32), "v"(ga) : "memory");
#endif
}
template <int N>
__device__ __forceinline__ void wait_async() {
#if __has_builtin(__builtin_amdgcn_s_wait_asynccnt)
  __builtin_amdgcn_s_wait_asynccnt(N);
#else
  asm volatile("s_wait_asynccnt %0" :: "i"(N) : "memory");
#endif
}

// ---------------- prep kernels ----------------
__global__ void mfl_cvt_x_f16(const float4* __restrict__ x, uint2* __restrict__ o, int n4) {
  int i = blockIdx.x * blockDim.x + threadIdx.x;
  if (i >= n4) return;
  float4 v = x[i];
  union { _Float16 h[4]; uint2 u; } r;
  r.h[0] = (_Float16)v.x; r.h[1] = (_Float16)v.y;
  r.h[2] = (_Float16)v.z; r.h[3] = (_Float16)v.w;
  o[i] = r.u;
}

__global__ void mfl_cvt_w_f16(const float4* __restrict__ w, uint2* __restrict__ o, int n4) {
  int i = blockIdx.x * blockDim.x + threadIdx.x;
  if (i >= n4) return;
  float4 v = w[i];
  union { _Float16 h[4]; uint2 u; } r;
  r.h[0] = (_Float16)quant_w_val(v.x); r.h[1] = (_Float16)quant_w_val(v.y);
  r.h[2] = (_Float16)quant_w_val(v.z); r.h[3] = (_Float16)quant_w_val(v.w);
  o[i] = r.u;
}

__global__ void mfl_cvt_bias(const float* __restrict__ b, float* __restrict__ o, int n) {
  int i = blockIdx.x * blockDim.x + threadIdx.x;
  if (i < n) o[i] = fp8_round(b[i], 3, -6);   // e4m3fn, back as f32
}

// ---------------- main WMMA GEMM: out = Xf16 * Wf16^T + bias ----------------
__global__ __launch_bounds__(256) void mfl_gemm_wmma(
    const _Float16* __restrict__ xf,   // [MD, KD]
    const _Float16* __restrict__ wf,   // [ND, KD]  (row n = weights for out-feature n)
    const float*    __restrict__ bias, // [ND]
    float*          __restrict__ out)  // [MD, ND]
{
  __shared__ __align__(16) unsigned char smem[2 * BUF_B];   // 40 KB double buffer

  const int tid  = threadIdx.x;
  const int lane = tid & 31;
  const int wid  = tid >> 5;           // 8 waves
  const int l15  = lane & 15;
  const int hi   = lane >> 4;          // half-wave select

  // ---- loader mapping: each thread owns four 16B chunks of the 256-row A|B tile
  const int cb = (lane & 3) * 16;                    // byte column within a 64B K-row
  const unsigned rbase = (unsigned)(wid * 32 + (lane >> 2));
  const _Float16* gsrc[4];
  unsigned ldsoff[4];
#pragma unroll
  for (int i = 0; i < 4; ++i) {
    unsigned R = rbase + i * 8;                      // 0..255, unique per (wid,i,lane/4)
    if (R < 128) {                                   // A rows (x tile)
      size_t row = (size_t)blockIdx.y * BM + R;
      gsrc[i]   = xf + row * KD + (cb >> 1);
      ldsoff[i] = R * PITCH + cb;
    } else {                                         // B rows (weight tile, N-major)
      unsigned r = R - 128;
      size_t row = (size_t)blockIdx.x * BN + r;
      gsrc[i]   = wf + row * KD + (cb >> 1);
      ldsoff[i] = (unsigned)TILE_B + r * PITCH + cb;
    }
  }

  // ---- per-wave compute tile: 64(M) x 32(N) -> 4x2 WMMA 16x16 tiles
  const int wm = wid >> 2;             // 0..1
  const int wn = wid & 3;              // 0..3
  unsigned aoff[4], boff[2];
#pragma unroll
  for (int f = 0; f < 4; ++f)          // A 16x32 f16: halves split K 0-7 / 8-15 (+16)
    aoff[f] = (unsigned)((wm * 64 + f * 16 + l15) * PITCH + hi * 16);
#pragma unroll
  for (int g = 0; g < 2; ++g)          // B 32x16 f16: halves split K 0-7 / 8-15 (+16)
    boff[g] = (unsigned)(TILE_B + (wn * 32 + g * 16 + l15) * PITCH + hi * 16);

  v8f acc[4][2];
#pragma unroll
  for (int f = 0; f < 4; ++f)
#pragma unroll
    for (int g = 0; g < 2; ++g)
      acc[f][g] = (v8f){0.f, 0.f, 0.f, 0.f, 0.f, 0.f, 0.f, 0.f};

  // prologue: tile 0 -> buffer 0
#pragma unroll
  for (int i = 0; i < 4; ++i)
    async_copy16(gsrc[i], smem + ldsoff[i]);

  for (int kt = 0; kt < NT; ++kt) {
    const int buf = kt & 1;
    if (kt + 1 < NT) {                 // prefetch next tile into other buffer
      const unsigned nb = (unsigned)((buf ^ 1) * BUF_B);
#pragma unroll
      for (int i = 0; i < 4; ++i)
        async_copy16(gsrc[i] + (size_t)(kt + 1) * BK, smem + nb + ldsoff[i]);
      wait_async<4>();                 // tile kt complete; kt+1 still in flight
    } else {
      wait_async<0>();
    }
    __syncthreads();

    const unsigned char* sb = smem + buf * BUF_B;
    union { v16h v; uint4 q[2]; } A[4], B[2];
#pragma unroll
    for (int f = 0; f < 4; ++f) {
      A[f].q[0] = *(const uint4*)(sb + aoff[f]);
      A[f].q[1] = *(const uint4*)(sb + aoff[f] + 32);
    }
#pragma unroll
    for (int g = 0; g < 2; ++g) {
      B[g].q[0] = *(const uint4*)(sb + boff[g]);
      B[g].q[1] = *(const uint4*)(sb + boff[g] + 32);
    }
#pragma unroll
    for (int f = 0; f < 4; ++f)
#pragma unroll
      for (int g = 0; g < 2; ++g)
        acc[f][g] = __builtin_amdgcn_wmma_f32_16x16x32_f16(
            false, A[f].v, false, B[g].v, (short)0, acc[f][g], false, false);

    __syncthreads();                   // buf free for the tile issued two steps ahead
  }

  // ---- epilogue: C layout lanes 0-15 -> M=v, lanes 16-31 -> M=v+8 ; N = lane&15
  const int mbase = (int)blockIdx.y * BM + wm * 64 + hi * 8;
#pragma unroll
  for (int g = 0; g < 2; ++g) {
    const int col = (int)blockIdx.x * BN + wn * 32 + g * 16 + l15;
    const float bv = bias[col];
#pragma unroll
    for (int f = 0; f < 4; ++f) {
#pragma unroll
      for (int v = 0; v < 8; ++v) {
        const int m = mbase + f * 16 + v;
        out[(size_t)m * ND + col] = acc[f][g][v] + bv;
      }
    }
  }
}

// ---------------- host launcher ----------------
extern "C" void kernel_launch(void* const* d_in, const int* in_sizes, int n_in,
                              void* d_out, int out_size, void* d_ws, size_t ws_size,
                              hipStream_t stream) {
  const float* x = (const float*)d_in[0];   // [4,2048,4096]
  const float* w = (const float*)d_in[1];   // [4096,4096]
  const float* b = (const float*)d_in[2];   // [4096]
  float* out = (float*)d_out;

  unsigned char* ws = (unsigned char*)d_ws;
  _Float16* xf = (_Float16*)ws;                                   // 67,108,864 B
  _Float16* wf = (_Float16*)(ws + (size_t)MD * KD * 2);           // 33,554,432 B
  float*    bf = (float*)(ws + (size_t)MD * KD * 2 + (size_t)ND * KD * 2); // 16 KB

  const int nx4 = MD * KD / 4;
  mfl_cvt_x_f16<<<(nx4 + 255) / 256, 256, 0, stream>>>((const float4*)x, (uint2*)xf, nx4);
  const int nw4 = ND * KD / 4;
  mfl_cvt_w_f16<<<(nw4 + 255) / 256, 256, 0, stream>>>((const float4*)w, (uint2*)wf, nw4);
  mfl_cvt_bias<<<(ND + 255) / 256, 256, 0, stream>>>(b, bf, ND);

  dim3 grid(ND / BN, MD / BM);   // 32 x 64
  mfl_gemm_wmma<<<grid, 256, 0, stream>>>(xf, wf, bf, out);
}
